// W8A8B8O8Linear_75290776699096
// MI455X (gfx1250) — 8-bit matrix operand order repaired and checked by reading, compiled, not run on hardware
//
#include <hip/hip_runtime.h>
#include <hip/hip_bf16.h>

typedef __attribute__((ext_vector_type(8))) int          v8i;
typedef __attribute__((ext_vector_type(4))) unsigned int u32x4;
typedef __attribute__((ext_vector_type(4))) int          i32x4;
typedef __attribute__((ext_vector_type(8))) int          i32x8;

#define TILE_M 128
#define TILE_N 128
#define TILE_K 128

#define LDS_ROW    144
#define TILE_LDS   (TILE_M * LDS_ROW)
#define BUF_LDS    (2 * TILE_LDS)

__global__ __launch_bounds__(256) void pack_i32_to_i8_kernel(
    const int* __restrict__ in, int* __restrict__ out, int n4) {
  int i = blockIdx.x * blockDim.x + threadIdx.x;
  if (i >= n4) return;
  const int4 v = *(const int4*)(in + (size_t)i * 4);
  unsigned p = ((unsigned)v.x & 0xFFu)
             | (((unsigned)v.y & 0xFFu) << 8)
             | (((unsigned)v.z & 0xFFu) << 16)
             | (((unsigned)v.w & 0xFFu) << 24);
  out[i] = (int)p;
}

__device__ __forceinline__ void tdm_load_tile_2d(
    unsigned lds_addr, const void* gptr,
    unsigned tile_w, unsigned tile_h,
    unsigned tensor_w, unsigned tensor_h, unsigned row_stride) {
  const unsigned long long ga = (unsigned long long)(size_t)gptr;
  u32x4 g0;
  g0[0] = 1u;
  g0[1] = lds_addr;
  g0[2] = (unsigned)ga;
  g0[3] = (unsigned)((ga >> 32) & 0x01FFFFFFu) | (2u << 30);
  i32x8 g1;
  g1[0] = (int)((1u << 20) | (4u << 22) | (3u << 25));
  g1[1] = (int)((tensor_w & 0xFFFFu) << 16);
  g1[2] = (int)(((tensor_w >> 16) & 0xFFFFu) | ((tensor_h & 0xFFFFu) << 16));
  g1[3] = (int)(((tensor_h >> 16) & 0xFFFFu) | ((tile_w & 0xFFFFu) << 16));
  g1[4] = (int)(tile_h & 0xFFFFu);
  g1[5] = (int)row_stride;
  g1[6] = 0;
  g1[7] = 0;
  const i32x4 z4 = {0, 0, 0, 0};
#if __clang_major__ >= 23
  const i32x8 z8 = {0, 0, 0, 0, 0, 0, 0, 0};
  __builtin_amdgcn_tensor_load_to_lds(g0, g1, z4, z4, z8, 0);
#else
  __builtin_amdgcn_tensor_load_to_lds(g0, g1, z4, z4, 0);
#endif
}

__global__ __launch_bounds__(256) void w8a8_gemm_wmma_kernel(
    const int* __restrict__ Apk, const int* __restrict__ Bpk,
    const int* __restrict__ bias,
    const float* __restrict__ p_alpha, const float* __restrict__ p_beta,
    float* __restrict__ out, int M, int N, int K) {

  __shared__ __align__(16) char smem[2 * BUF_LDS];

  const int tid  = threadIdx.x;
  const int lane = tid & 31;
  const int wave = tid >> 5;
  const int wm   = wave & 3;
  const int wn   = wave >> 2;

  const int n0 = blockIdx.x * TILE_N;
  const int m0 = blockIdx.y * TILE_M;

  const int half = lane >> 4;
  const int l15  = lane & 15;

  const char* Agbase = (const char*)Apk + (size_t)m0 * K;
  const char* Bgbase = (const char*)Bpk + (size_t)n0 * K;
  const unsigned ldsBase = (unsigned)(size_t)&smem[0];

  v8i acc[2][4];
#pragma unroll
  for (int i = 0; i < 2; ++i)
#pragma unroll
    for (int j = 0; j < 4; ++j)
      acc[i][j] = (v8i)(0);

  const int ksteps = K / TILE_K;

  if (wave == 0) {
    tdm_load_tile_2d(ldsBase,            Agbase, TILE_K, TILE_M,
                     (unsigned)K, (unsigned)M, (unsigned)K);
    tdm_load_tile_2d(ldsBase + TILE_LDS, Bgbase, TILE_K, TILE_N,
                     (unsigned)K, (unsigned)N, (unsigned)K);
  }

  for (int it = 0; it < ksteps; ++it) {
    __builtin_amdgcn_s_wait_tensorcnt(0);
    __syncthreads();

    if (wave == 0 && (it + 1) < ksteps) {
      const unsigned nb = (unsigned)((it + 1) & 1) * BUF_LDS;
      const int k1 = (it + 1) * TILE_K;
      tdm_load_tile_2d(ldsBase + nb,            Agbase + k1, TILE_K, TILE_M,
                       (unsigned)K, (unsigned)M, (unsigned)K);
      tdm_load_tile_2d(ldsBase + nb + TILE_LDS, Bgbase + k1, TILE_K, TILE_N,
                       (unsigned)K, (unsigned)N, (unsigned)K);
    }

    const char* As = smem + (it & 1) * BUF_LDS;
    const char* Bs = As + TILE_LDS;

#pragma unroll
    for (int kk = 0; kk < TILE_K; kk += 64) {
      v8i afrag[2];
#pragma unroll
      for (int i = 0; i < 2; ++i) {
        const char* base = As + (wm * 32 + i * 16 + l15) * LDS_ROW + kk;
        const int2 p0 = *(const int2*)(base + half * 16);
        const int2 p1 = *(const int2*)(base + 8 + half * 16);
        const int2 p2 = *(const int2*)(base + 32 + half * 16);
        const int2 p3 = *(const int2*)(base + 40 + half * 16);
        v8i a = {p0.x, p0.y, p1.x, p1.y, p2.x, p2.y, p3.x, p3.y};
        afrag[i] = a;
      }
      v8i bfrag[4];
#pragma unroll
      for (int j = 0; j < 4; ++j) {
        const char* base = Bs + (wn * 64 + j * 16 + l15) * LDS_ROW + kk;
        const int4 q0 = *(const int4*)(base + half * 16);
        const int4 q1 = *(const int4*)(base + 32 + half * 16);
        v8i b = {q0.x, q0.y, q0.z, q0.w, q1.x, q1.y, q1.z, q1.w};
        bfrag[j] = b;
      }
#pragma unroll
      for (int i = 0; i < 2; ++i)
#pragma unroll
        for (int j = 0; j < 4; ++j)
          acc[i][j] = __builtin_amdgcn_wmma_i32_16x16x64_iu8(
               true, afrag[i],  true, bfrag[j],
              acc[i][j],  false,  false);
    }
  }

  const float alpha = *p_alpha;
  const float beta  = *p_beta;

#pragma unroll
  for (int i = 0; i < 2; ++i) {
#pragma unroll
    for (int j = 0; j < 4; ++j) {
      const int colN = n0 + wn * 64 + j * 16 + l15;
      const float bv = beta * (float)bias[colN];
#pragma unroll
      for (int r = 0; r < 8; ++r) {
        const int rowM = m0 + wm * 32 + i * 16 + r + half * 8;
        float y = fmaf(alpha, (float)acc[i][j][r], bv);
        y = rintf(y);
        y = fminf(127.0f, fmaxf(-128.0f, y));
        out[(size_t)rowM * N + colN] = y;
      }
    }
  }
}

extern "C" void kernel_launch(void* const* d_in, const int* in_sizes, int n_in,
                              void* d_out, int out_size, void* d_ws, size_t ws_size,
                              hipStream_t stream) {
  const int*   x    = (const int*)d_in[0];
  const int*   w    = (const int*)d_in[1];
  const int*   bias = (const int*)d_in[2];
  const float* pa   = (const float*)d_in[3];
  const float* pb   = (const float*)d_in[4];
  float*       out  = (float*)d_out;

  const int N = in_sizes[2];
  const int K = in_sizes[1] / N;
  const int M = in_sizes[0] / K;

  int* Apk = (int*)d_ws;
  int* Bpk = Apk + ((size_t)M * K) / 4;

  const int nA4 = M * (K / 4);
  const int nB4 = N * (K / 4);
  pack_i32_to_i8_kernel<<<(nA4 + 255) / 256, 256, 0, stream>>>(x, Apk, nA4);
  pack_i32_to_i8_kernel<<<(nB4 + 255) / 256, 256, 0, stream>>>(w, Bpk, nB4);

  dim3 grid(N / TILE_N, M / TILE_M);
  w8a8_gemm_wmma_kernel<<<grid, 256, 0, stream>>>(Apk, Bpk, bias, pa, pb,
                                                  out, M, N, K);
}
